// NodeModel_49503793053940
// MI455X (gfx1250) — matrix operand order repaired and checked by reading, not compiled, not run on hardware
//
#include <hip/hip_runtime.h>

#define NN 50000
#define NE 800000
#define NODE_IN 128
#define NODE_OUT 128
#define EOUT 64
#define HID 512
#define DIN 320  // NODE_IN + 3*EOUT

typedef __attribute__((ext_vector_type(16))) __bf16 v16bf;
typedef __attribute__((ext_vector_type(8)))  float  v8f;

// ---------- helpers ----------
static __device__ __forceinline__ unsigned short f32_to_bf16(float f) {
  unsigned u = __float_as_uint(f);
  u += 0x7FFFu + ((u >> 16) & 1u);   // round-to-nearest-even
  return (unsigned short)(u >> 16);
}
// monotone float <-> uint key (for atomicMax on floats incl. negatives)
static __device__ __forceinline__ unsigned keyf(float f) {
  unsigned u = __float_as_uint(f);
  return (u & 0x80000000u) ? ~u : (u | 0x80000000u);
}
static __device__ __forceinline__ float unkeyf(unsigned k) {
  unsigned u = (k & 0x80000000u) ? (k & 0x7FFFFFFFu) : ~k;
  return __uint_as_float(u);
}
#define NEG_INF_KEY 0x007FFFFFu  // keyf(-inf)

// CDNA5 async global->LDS copy (ASYNCcnt-tracked), 16B per lane.
// LDS dest VGPR holds LDS byte offset = low 32 bits of the generic pointer
// (LDS aperture low dword is the offset per ISA 10.2).
static __device__ __forceinline__ void async_b128(void* lds, const void* g) {
  unsigned loff = (unsigned)(unsigned long long)(uintptr_t)lds;
  unsigned long long ga = (unsigned long long)(uintptr_t)g;
  asm volatile("global_load_async_to_lds_b128 %0, %1, off"
               :: "v"(loff), "v"(ga) : "memory");
}
static __device__ __forceinline__ void wait_async_le3() {
  asm volatile("s_wait_asynccnt 0x3" ::: "memory");
}
static __device__ __forceinline__ void wait_async_0() {
  asm volatile("s_wait_asynccnt 0x0" ::: "memory");
}

// ---------- 1) init aggregates ----------
__global__ void init_kernel(float* s_sum, unsigned* s_maxk, float* cnt) {
  int i = blockIdx.x * blockDim.x + threadIdx.x;
  int total = NN * EOUT;
  for (; i < total; i += gridDim.x * blockDim.x) {
    s_sum[i]  = 0.0f;
    s_maxk[i] = NEG_INF_KEY;
    if (i < NN) cnt[i] = 0.0f;
  }
}

// ---------- 2) scatter sum/max/count ----------
__global__ void scatter_kernel(const float* __restrict__ edge_attr,
                               const int*   __restrict__ edge_index,
                               float* s_sum, unsigned* s_maxk, float* cnt) {
  long long i = (long long)blockIdx.x * blockDim.x + threadIdx.x;
  if (i >= (long long)NE * 16) return;          // 16 threads/edge, 4 feats each
  int e  = (int)(i >> 4);
  int c4 = (int)(i & 15) * 4;
  int col = edge_index[NE + e];                 // row 1 of edge_index
  float4 v = *(const float4*)(edge_attr + (size_t)e * EOUT + c4);
  size_t base = (size_t)col * EOUT + c4;
  atomicAdd(&s_sum[base + 0], v.x);
  atomicAdd(&s_sum[base + 1], v.y);
  atomicAdd(&s_sum[base + 2], v.z);
  atomicAdd(&s_sum[base + 3], v.w);
  atomicMax(&s_maxk[base + 0], keyf(v.x));
  atomicMax(&s_maxk[base + 1], keyf(v.y));
  atomicMax(&s_maxk[base + 2], keyf(v.z));
  atomicMax(&s_maxk[base + 3], keyf(v.w));
  if ((i & 15) == 0) atomicAdd(&cnt[col], 1.0f);
}

// ---------- 3) build h = [x | sum | max | mean] as bf16 ----------
__global__ void build_h_kernel(const float* __restrict__ x,
                               const float* __restrict__ s_sum,
                               const unsigned* __restrict__ s_maxk,
                               const float* __restrict__ cnt,
                               unsigned short* __restrict__ hA) {
  long long i = (long long)blockIdx.x * blockDim.x + threadIdx.x;
  if (i >= (long long)NN * DIN) return;
  int n = (int)(i / DIN);
  int c = (int)(i % DIN);
  float v;
  if (c < NODE_IN) {
    v = x[(size_t)n * NODE_IN + c];
  } else if (c < NODE_IN + EOUT) {
    v = s_sum[(size_t)n * EOUT + (c - NODE_IN)];
  } else if (c < NODE_IN + 2 * EOUT) {
    float m = unkeyf(s_maxk[(size_t)n * EOUT + (c - NODE_IN - EOUT)]);
    v = (__builtin_isinf(m) && m < 0.0f) ? 0.0f : m;   // empty segment -> 0
  } else {
    float s = s_sum[(size_t)n * EOUT + (c - NODE_IN - 2 * EOUT)];
    v = s / fmaxf(cnt[n], 1.0f);
  }
  hA[(size_t)n * DIN + c] = f32_to_bf16(v);
}

// ---------- 4) weights -> bf16, transposed to [N][K] ----------
__global__ void convert_w_kernel(const float* __restrict__ W1,
                                 const float* __restrict__ W2,
                                 unsigned short* __restrict__ w1t,
                                 unsigned short* __restrict__ w2t) {
  int i = blockIdx.x * blockDim.x + threadIdx.x;
  const int n1 = DIN * HID;                 // 163840
  const int n2 = HID * NODE_OUT;            // 65536
  if (i < n1) {
    int n = i % HID, k = i / HID;           // W1[k][n] row-major (DIN x HID)
    w1t[(size_t)n * DIN + k] = f32_to_bf16(W1[(size_t)k * HID + n]);
  } else if (i < n1 + n2) {
    int j = i - n1;
    int n = j % NODE_OUT, k = j / NODE_OUT; // W2[k][n] (HID x NODE_OUT)
    w2t[(size_t)n * HID + k] = f32_to_bf16(W2[(size_t)k * NODE_OUT + n]);
  }
}

// ---------- 5) bf16 WMMA GEMM: C = act(A @ Bt^T + bias) ----------
// A: MxK row-major bf16, Bt: NxK row-major bf16 (B transposed), bias: N f32
// Block tile 64(M) x 128(N); 8 waves (wave32): wm=wave>>1 in [0,4), wn=wave&1.
// Each wave: 16 rows x 64 cols = 4 x (16x16) WMMA accumulators.
// Tiles staged with double-buffered GLOBAL_LOAD_ASYNC_TO_LDS_B128 (ASYNCcnt).
#define APAD 40   // 32 + 8 ushorts -> 80B row stride, conflict-free ds_load_b128
template <bool RELU, bool OUT_BF16>
__global__ __launch_bounds__(256) void gemm_bf16_kernel(
    const unsigned short* __restrict__ A,
    const unsigned short* __restrict__ Bt,
    const float* __restrict__ bias,
    void* __restrict__ Cout, int M, int K, int N) {
  __shared__ unsigned short sA[2][64 * APAD];
  __shared__ unsigned short sB[2][128 * APAD];

  const int tid  = threadIdx.x;
  const int lane = tid & 31;
  const int wave = tid >> 5;
  const int wm   = wave >> 1;     // 0..3
  const int wn   = wave & 1;      // 0..1
  const int hi   = lane >> 4;     // 0/1 (half-wave)
  const int l16  = lane & 15;

  const int rowBase = blockIdx.x * 64;
  const int colBase = blockIdx.y * 128;

  v8f acc[4];
  const v8f vzero = {0.f, 0.f, 0.f, 0.f, 0.f, 0.f, 0.f, 0.f};
#pragma unroll
  for (int t = 0; t < 4; ++t) acc[t] = vzero;

  // cooperative staging indices (16B async copies, fully coalesced)
  const int ar = tid >> 2;            // 0..63 (A row in tile)
  const int ak = (tid & 3) * 8;       // 0,8,16,24
  const int arow = min(rowBase + ar, M - 1);   // clamp tail (masked at store)
  const int bn = tid >> 1;            // 0..127 (B row = output col)
  const int bk = (tid & 1) * 16;      // 0,16
  const int bng = colBase + bn;

  const int nsteps = K >> 5;  // K/32, K is a multiple of 32

  // each thread contributes 3 async b128 copies per stage
  auto issue_stage = [&](int step, int buf) {
    const int k0 = step * 32;
    async_b128(&sA[buf][ar * APAD + ak],     A  + (size_t)arow * K + k0 + ak);
    async_b128(&sB[buf][bn * APAD + bk],     Bt + (size_t)bng  * K + k0 + bk);
    async_b128(&sB[buf][bn * APAD + bk + 8], Bt + (size_t)bng  * K + k0 + bk + 8);
  };

  issue_stage(0, 0);
  if (nsteps > 1) issue_stage(1, 1);

  for (int i = 0; i < nsteps; ++i) {
    const int buf = i & 1;
    // stage i's 3 asyncs are the oldest; stage i+1's (if issued) may remain
    if (i + 1 < nsteps) wait_async_le3();
    else                wait_async_0();
    __syncthreads();   // all waves' stage-i data visible in LDS

    // A fragment (ISA 16-bit A 16x32 layout): lane<16 -> K[0..7],[16..23]
    union { v16bf v; uint4 q[2]; } af;
    const int arl = wm * 16 + l16;
    af.q[0] = *(const uint4*)(&sA[buf][arl * APAD + hi * 8]);
    af.q[1] = *(const uint4*)(&sA[buf][arl * APAD + 16 + hi * 8]);
    // preload all 4 B fragments (col N = lane%16, K = hi*8+[0..7], 16+hi*8+[0..7])
    union { v16bf v; uint4 q[2]; } bf[4];
#pragma unroll
    for (int t = 0; t < 4; ++t) {
      const int bc = wn * 64 + t * 16 + l16;
      bf[t].q[0] = *(const uint4*)(&sB[buf][bc * APAD + hi * 8]);
      bf[t].q[1] = *(const uint4*)(&sB[buf][bc * APAD + 16 + hi * 8]);
    }
#pragma unroll
    for (int t = 0; t < 4; ++t) {
      acc[t] = __builtin_amdgcn_wmma_f32_16x16x32_bf16(
          false, af.v, false, bf[t].v, (short)0, acc[t], false, false);
    }
    __syncthreads();   // all waves done reading buf before overwrite
    if (i + 2 < nsteps) issue_stage(i + 2, buf);
  }

  // writeback: C layout lane L: col = L%16, VGPR r -> row r + 8*(L>=16)
  const int mBase = rowBase + wm * 16 + hi * 8;
#pragma unroll
  for (int t = 0; t < 4; ++t) {
    const int n = colBase + wn * 64 + t * 16 + l16;
    const float bv = bias[n];
#pragma unroll
    for (int r = 0; r < 8; ++r) {
      const int m = mBase + r;
      if (m < M) {
        float v = acc[t][r] + bv;
        if (RELU) v = v > 0.0f ? v : 0.0f;
        if (OUT_BF16)
          ((unsigned short*)Cout)[(size_t)m * N + n] = f32_to_bf16(v);
        else
          ((float*)Cout)[(size_t)m * N + n] = v;
      }
    }
  }
}

// ---------- host ----------
extern "C" void kernel_launch(void* const* d_in, const int* in_sizes, int n_in,
                              void* d_out, int out_size, void* d_ws, size_t ws_size,
                              hipStream_t stream) {
  const float* x          = (const float*)d_in[0];
  const int*   edge_index = (const int*)  d_in[1];
  const float* edge_attr  = (const float*)d_in[2];
  // d_in[3] = u, d_in[4] = batch : unused by the reference node model
  const float* W1 = (const float*)d_in[5];
  const float* b1 = (const float*)d_in[6];
  const float* W2 = (const float*)d_in[7];
  const float* b2 = (const float*)d_in[8];
  float* out = (float*)d_out;

  // carve workspace (aligned 256B)
  size_t off = 0;
  auto carve = [&](size_t bytes) -> void* {
    void* p = (char*)d_ws + off;
    off = (off + bytes + 255) & ~(size_t)255;
    return p;
  };
  float*          s_sum  = (float*)         carve((size_t)NN * EOUT * 4);
  unsigned*       s_maxk = (unsigned*)      carve((size_t)NN * EOUT * 4);
  float*          cnt    = (float*)         carve((size_t)NN * 4);
  unsigned short* hA     = (unsigned short*)carve((size_t)NN * DIN * 2);
  unsigned short* h2     = (unsigned short*)carve((size_t)NN * HID * 2);
  unsigned short* w1t    = (unsigned short*)carve((size_t)DIN * HID * 2);
  unsigned short* w2t    = (unsigned short*)carve((size_t)HID * NODE_OUT * 2);

  // 1) init aggregates
  {
    int total = NN * EOUT;
    init_kernel<<<(total + 255) / 256, 256, 0, stream>>>(s_sum, s_maxk, cnt);
  }
  // 2) scatter
  {
    long long total = (long long)NE * 16;
    scatter_kernel<<<(unsigned)((total + 255) / 256), 256, 0, stream>>>(
        edge_attr, edge_index, s_sum, s_maxk, cnt);
  }
  // 3) build h (bf16)
  {
    long long total = (long long)NN * DIN;
    build_h_kernel<<<(unsigned)((total + 255) / 256), 256, 0, stream>>>(
        x, s_sum, s_maxk, cnt, hA);
  }
  // 4) weights -> bf16 transposed
  {
    int total = DIN * HID + HID * NODE_OUT;
    convert_w_kernel<<<(total + 255) / 256, 256, 0, stream>>>(W1, W2, w1t, w2t);
  }
  // 5) layer 1: h2 = relu(h @ W1 + b1)   [50000x320 @ 320x512]
  {
    dim3 grid((NN + 63) / 64, HID / 128);
    gemm_bf16_kernel<true, true><<<grid, 256, 0, stream>>>(
        hA, w1t, b1, (void*)h2, NN, DIN, HID);
  }
  // 6) layer 2: out = h2 @ W2 + b2       [50000x512 @ 512x128]
  {
    dim3 grid((NN + 63) / 64, NODE_OUT / 128);
    gemm_bf16_kernel<false, false><<<grid, 256, 0, stream>>>(
        h2, w2t, b2, (void*)out, NN, HID, NODE_OUT);
  }
}
